// PNANet_30296699306196
// MI455X (gfx1250) — matrix operand order repaired and checked by reading, not compiled, not run on hardware
//
#include <hip/hip_runtime.h>
#include <hip/hip_bf16.h>
#include <math.h>

typedef _Float16 half_t;
typedef __attribute__((ext_vector_type(16))) _Float16 v16h;
typedef __attribute__((ext_vector_type(8)))  float    v8f;

#define NN    20000
#define EE    320000
#define DD    128
#define LL    4
#define GG    256
#define AVG_LOG 2.833213344056216f
#define EPSV  1e-5f

static __device__ __forceinline__ v8f wmma16(v16h a, v16h b, v8f c) {
  return __builtin_amdgcn_wmma_f32_16x16x32_f16(false, a, false, b, (short)0, c, false, false);
}

// monotone float<->uint map so atomicMax/Min on uint == float max/min
static __device__ __forceinline__ unsigned mapf(float f) {
  unsigned u = __float_as_uint(f);
  return (u & 0x80000000u) ? ~u : (u | 0x80000000u);
}
static __device__ __forceinline__ float unmapf(unsigned u) {
  u = (u & 0x80000000u) ? (u & 0x7FFFFFFFu) : ~u;
  return __uint_as_float(u);
}

// ---------------- utility kernels ----------------
__global__ void fill_u32_kernel(unsigned* p, unsigned v, int n) {
  int i = blockIdx.x * blockDim.x + threadIdx.x;
  if (i < n) p[i] = v;
}
// one interleaved cell per (node,col): {sum.f32, sumsq.f32, maxbits.u32, minbits.u32}
__global__ void agg_init_kernel(uint4* aggbuf, int n) {
  int i = blockIdx.x * blockDim.x + threadIdx.x;
  if (i >= n) return;
  aggbuf[i] = make_uint4(0u, 0u, 0u, 0xFFFFFFFFu);
}
// pack f32 row-major weight [K x 128] into WMMA B-fragment order:
// out[(((kk*8 + ntile)*32 + lane)*16 + j)] = W[(kk*32 + (lane<16?0:8) + (j>>3)*16 + (j&7))*128 + ntile*16 + lane%16]
// -> each lane's fragment is one contiguous 32B v16h load in the GEMM.
__global__ void pack_w_kernel(const float* __restrict__ W, half_t* __restrict__ out,
                              int K, int nLayers) {
  int idx = blockIdx.x * blockDim.x + threadIdx.x;
  int perLayer = K * 128;
  if (idx >= nLayers * perLayer) return;
  int layer = idx / perLayer;
  int r = idx - layer * perLayer;
  int j    = r & 15;
  int lane = (r >> 4) & 31;
  int w    = (r >> 9) & 7;
  int kk   = r >> 12;
  int row = kk * 32 + ((lane < 16) ? 0 : 8) + (j >> 3) * 16 + (j & 7);
  int col = w * 16 + (lane & 15);
  out[idx] = (half_t)W[(size_t)layer * perLayer + (size_t)row * 128 + col];
}
__global__ void node_embed_kernel(const int* __restrict__ x, const float* __restrict__ emb,
                                  float* __restrict__ h, half_t* __restrict__ hf) {
  int i = blockIdx.x * blockDim.x + threadIdx.x;
  if (i >= NN * DD) return;
  int node = i >> 7, col = i & 127;
  float v = emb[x[node] * DD + col];
  h[i] = v; hf[i] = (half_t)v;
}
__global__ void edge_embed_kernel(const int* __restrict__ attr, const float* __restrict__ emb,
                                  half_t* __restrict__ ef) {
  int i = blockIdx.x * blockDim.x + threadIdx.x;
  if (i >= EE * DD) return;
  int e = i >> 7, col = i & 127;
  ef[i] = (half_t)emb[attr[e] * DD + col];
}
__global__ void deg_kernel(const int* __restrict__ dst, float* __restrict__ deg) {
  int e = blockIdx.x * blockDim.x + threadIdx.x;
  if (e < EE) atomicAdd(&deg[dst[e]], 1.0f);
}
__global__ void scaler_kernel(const float* __restrict__ deg, float* __restrict__ amp,
                              float* __restrict__ att) {
  int i = blockIdx.x * blockDim.x + threadIdx.x;
  if (i >= NN) return;
  float logd = log1pf(deg[i]);
  amp[i] = logd / AVG_LOG;
  att[i] = AVG_LOG / fmaxf(logd, EPSV);
}

// ---------------- edge pretrans GEMM + fused scatter aggregation ----------------
// grid: E/16 tiles of 16 edges, 256 threads = 8 waves; wave w owns output cols [16w,16w+16)
__global__ __launch_bounds__(256) void pretrans_agg_kernel(
    const half_t* __restrict__ hf, const half_t* __restrict__ ef,
    const half_t* __restrict__ Wpk,       // packed B fragments, 12 ktiles
    const float* __restrict__ bias,       // [128]
    const int* __restrict__ srcIdx, const int* __restrict__ dstIdx,
    uint4* __restrict__ aggbuf) {
  __shared__ __align__(16) half_t Asm[16 * 384];
  __shared__ int dsm[16], ssm[16];
  const int e0 = blockIdx.x * 16;
  const int tid = threadIdx.x;
  if (tid < 16) dsm[tid] = dstIdx[e0 + tid];
  else if (tid < 32) ssm[tid - 16] = srcIdx[e0 + tid - 16];
  __syncthreads();
  // stage A = cat(h[dst], h[src], e) : 16 rows x 384 halves, 16B chunks
  for (int c = tid; c < 768; c += 256) {
    int row = c / 48, off = (c % 48) * 8;
    const half_t* sp;
    if (off < 128)       sp = hf + (size_t)dsm[row] * DD + off;
    else if (off < 256)  sp = hf + (size_t)ssm[row] * DD + (off - 128);
    else                 sp = ef + (size_t)(e0 + row) * DD + (off - 256);
    *(uint4*)&Asm[row * 384 + off] = *(const uint4*)sp;
  }
  __syncthreads();
  const int wave = tid >> 5, lane = tid & 31;
  const int arow = lane & 15;
  const int kaOff = (lane < 16) ? 0 : 8;
  const v16h* __restrict__ Wv = (const v16h*)Wpk;
  const int fbase = wave * 32 + lane;
  v8f acc = {};
  for (int kk = 0; kk < 12; ++kk) {
    const int k0 = kk * 32;
    __builtin_prefetch(&Wv[(kk + 1) * 256 + fbase], 0, 1);
    v16h a;
#pragma unroll
    for (int j = 0; j < 16; ++j)
      a[j] = Asm[arow * 384 + k0 + kaOff + (j >> 3) * 16 + (j & 7)];
    v16h b = Wv[kk * 256 + fbase];
    acc = wmma16(a, b, acc);
  }
  const int col = wave * 16 + (lane & 15);
  const float bv = bias[col];
#pragma unroll
  for (int v = 0; v < 8; ++v) {
    int m = v + ((lane < 16) ? 0 : 8);
    float val = acc[v] + bv;
    val = (val > 0.f) ? val : 0.01f * val;       // LeakyReLU
    // one address for all four aggregators: +0/+4/+8/+12 fold into imm offsets
    unsigned* cell = (unsigned*)&aggbuf[(size_t)dsm[m] * DD + col];
    atomicAdd((float*)cell + 0, val);
    atomicAdd((float*)cell + 1, val * val);
    unsigned ub = mapf(val);
    atomicMax(cell + 2, ub);
    atomicMin(cell + 3, ub);
  }
}

// ---------------- per-node aggregator finalize -> f16 [N,512] = [mean,max,min,std] ----------------
__global__ void agg_finalize_kernel(const uint4* __restrict__ aggbuf,
                                    const float* __restrict__ deg, half_t* __restrict__ aggf) {
  int i = blockIdx.x * blockDim.x + threadIdx.x;
  if (i >= NN * DD) return;
  int node = i >> 7, col = i & 127;
  uint4 cell = aggbuf[i];                          // single b128 load
  float d = deg[node];
  float dsafe = fmaxf(d, 1.0f);
  bool nb = d > 0.0f;
  float mean = __uint_as_float(cell.x) / dsafe;
  float msq  = __uint_as_float(cell.y) / dsafe;
  float sd = nb ? sqrtf(fmaxf(msq - mean * mean, 0.0f) + EPSV) : 0.0f;
  float mx = nb ? unmapf(cell.z) : 0.0f;
  float mn = nb ? unmapf(cell.w) : 0.0f;
  size_t base = (size_t)node * 512;
  aggf[base + col]       = (half_t)mean;
  aggf[base + 128 + col] = (half_t)mx;
  aggf[base + 256 + col] = (half_t)mn;
  aggf[base + 384 + col] = (half_t)sd;
}

// ---------------- posttrans: h@W0 + A@W1 + amp*(A@W2) + att*(A@W3) ----------------
// packed post_W: ktiles 0..3 = W0 (h), 4..19 = W1, 20..35 = W2, 36..51 = W3
__global__ __launch_bounds__(256) void posttrans_kernel(
    const half_t* __restrict__ hf, const half_t* __restrict__ aggf,
    const half_t* __restrict__ Wpk,
    const float* __restrict__ bias, const float* __restrict__ amp, const float* __restrict__ att,
    half_t* __restrict__ ph1) {
  __shared__ __align__(16) half_t Hsm[16 * 128];
  __shared__ __align__(16) half_t Asm[16 * 512];
  const int n0node = blockIdx.x * 16;
  const int tid = threadIdx.x;
  { int row = tid / 16, off = (tid % 16) * 8;
    *(uint4*)&Hsm[row * 128 + off] = *(const uint4*)&hf[(size_t)(n0node + row) * DD + off]; }
  for (int c = tid; c < 1024; c += 256) {
    int row = c / 64, off = (c % 64) * 8;
    *(uint4*)&Asm[row * 512 + off] = *(const uint4*)&aggf[(size_t)(n0node + row) * 512 + off];
  }
  __syncthreads();
  const int wave = tid >> 5, lane = tid & 31;
  const int arow = lane & 15;
  const int kaOff = (lane < 16) ? 0 : 8;
  const v16h* __restrict__ Wv = (const v16h*)Wpk;
  const int fbase = wave * 32 + lane;
  v8f ch = {}, c1 = {}, c2 = {}, c3 = {};
#pragma unroll
  for (int kk = 0; kk < 4; ++kk) {              // h @ Wp[0:128]
    const int k0 = kk * 32;
    v16h a;
#pragma unroll
    for (int j = 0; j < 16; ++j)
      a[j] = Hsm[arow * 128 + k0 + kaOff + (j >> 3) * 16 + (j & 7)];
    ch = wmma16(a, Wv[kk * 256 + fbase], ch);
  }
  for (int kk = 0; kk < 16; ++kk) {             // agg @ {W1,W2,W3}, one shared A fragment
    const int k0 = kk * 32;
    __builtin_prefetch(&Wv[(4 + kk + 1) * 256 + fbase], 0, 1);
    __builtin_prefetch(&Wv[(20 + kk + 1) * 256 + fbase], 0, 1);
    __builtin_prefetch(&Wv[(36 + kk + 1) * 256 + fbase], 0, 1);
    v16h a;
#pragma unroll
    for (int j = 0; j < 16; ++j)
      a[j] = Asm[arow * 512 + k0 + kaOff + (j >> 3) * 16 + (j & 7)];
    c1 = wmma16(a, Wv[(4  + kk) * 256 + fbase], c1);
    c2 = wmma16(a, Wv[(20 + kk) * 256 + fbase], c2);
    c3 = wmma16(a, Wv[(36 + kk) * 256 + fbase], c3);
  }
  const int col = wave * 16 + (lane & 15);
  const float bv = bias[col];
#pragma unroll
  for (int v = 0; v < 8; ++v) {
    int node = n0node + v + ((lane < 16) ? 0 : 8);
    float val = ch[v] + c1[v] + amp[node] * c2[v] + att[node] * c3[v] + bv;
    val = (val > 0.f) ? val : 0.01f * val;
    ph1[(size_t)node * DD + col] = (half_t)val;
  }
}

// ---------------- mix GEMM + graph_norm + BN partial sums ----------------
__global__ __launch_bounds__(256) void mix_kernel(
    const half_t* __restrict__ ph1, const half_t* __restrict__ Wpk,
    const float* __restrict__ bias, const float* __restrict__ snorm,
    float* __restrict__ ph, float* __restrict__ colsum, float* __restrict__ colsumsq) {
  __shared__ __align__(16) half_t Hsm[16 * 128];
  const int n0node = blockIdx.x * 16;
  const int tid = threadIdx.x;
  { int row = tid / 16, off = (tid % 16) * 8;
    *(uint4*)&Hsm[row * 128 + off] = *(const uint4*)&ph1[(size_t)(n0node + row) * DD + off]; }
  __syncthreads();
  const int wave = tid >> 5, lane = tid & 31;
  const int arow = lane & 15;
  const int kaOff = (lane < 16) ? 0 : 8;
  const v16h* __restrict__ Wv = (const v16h*)Wpk;
  const int fbase = wave * 32 + lane;
  v8f acc = {};
#pragma unroll
  for (int kk = 0; kk < 4; ++kk) {
    const int k0 = kk * 32;
    v16h a;
#pragma unroll
    for (int j = 0; j < 16; ++j)
      a[j] = Hsm[arow * 128 + k0 + kaOff + (j >> 3) * 16 + (j & 7)];
    acc = wmma16(a, Wv[kk * 256 + fbase], acc);
  }
  const int col = wave * 16 + (lane & 15);
  const float bv = bias[col];
  float psum = 0.f, psq = 0.f;
#pragma unroll
  for (int v = 0; v < 8; ++v) {
    int node = n0node + v + ((lane < 16) ? 0 : 8);
    float val = acc[v] + bv;
    val = (val > 0.f) ? val : 0.01f * val;
    val *= snorm[node];
    ph[(size_t)node * DD + col] = val;
    psum += val; psq += val * val;
  }
  // lane L and L+16 hold the same column -> combine then one atomic per col per block
  psum += __shfl_down(psum, 16);
  psq  += __shfl_down(psq, 16);
  if (lane < 16) {
    atomicAdd(&colsum[col], psum);
    atomicAdd(&colsumsq[col], psq);
  }
}

// ---------------- BN finalize + residual, refresh f16 copy ----------------
__global__ void bn_residual_kernel(float* __restrict__ h, half_t* __restrict__ hf,
                                   const float* __restrict__ ph,
                                   const float* __restrict__ colsum, const float* __restrict__ colsumsq,
                                   const float* __restrict__ gamma, const float* __restrict__ beta) {
  int i = blockIdx.x * blockDim.x + threadIdx.x;
  if (i >= NN * DD) return;
  int col = i & 127;
  float mu  = colsum[col] * (1.0f / NN);
  float var = colsumsq[col] * (1.0f / NN) - mu * mu;
  float val = h[i] + gamma[col] * (ph[i] - mu) * rsqrtf(fmaxf(var, 0.0f) + EPSV) + beta[col];
  h[i] = val; hf[i] = (half_t)val;
}

// ---------------- global mean pool + MLP readout ----------------
__global__ void pool_kernel(const float* __restrict__ h, const int* __restrict__ batch,
                            float* __restrict__ gsum, float* __restrict__ gcnt) {
  int i = blockIdx.x * blockDim.x + threadIdx.x;
  if (i >= NN * DD) return;
  int node = i >> 7, col = i & 127;
  int b = batch[node];
  atomicAdd(&gsum[(size_t)b * DD + col], h[i]);
  if (col == 0) atomicAdd(&gcnt[b], 1.0f);
}
__global__ void readout_kernel(const float* __restrict__ gsum, const float* __restrict__ gcnt,
                               const float* __restrict__ W1, const float* __restrict__ b1,
                               const float* __restrict__ W2, const float* __restrict__ b2,
                               const float* __restrict__ W3, const float* __restrict__ b3,
                               float* __restrict__ out) {
  __shared__ float g[128], t1[64], t2[32];
  int b = blockIdx.x, t = threadIdx.x;
  float c = fmaxf(gcnt[b], 1.0f);
  g[t] = gsum[(size_t)b * DD + t] / c;
  __syncthreads();
  if (t < 64) {
    float s = b1[t];
    for (int k = 0; k < 128; ++k) s += g[k] * W1[k * 64 + t];
    t1[t] = fmaxf(s, 0.0f);
  }
  __syncthreads();
  if (t < 32) {
    float s = b2[t];
    for (int k = 0; k < 64; ++k) s += t1[k] * W2[k * 32 + t];
    t2[t] = fmaxf(s, 0.0f);
  }
  __syncthreads();
  if (t == 0) {
    float s = b3[0];
    for (int k = 0; k < 32; ++k) s += t2[k] * W3[k];
    out[b] = s;
  }
}

static inline int cdiv(int a, int b) { return (a + b - 1) / b; }

extern "C" void kernel_launch(void* const* d_in, const int* in_sizes, int n_in,
                              void* d_out, int out_size, void* d_ws, size_t ws_size,
                              hipStream_t stream) {
  const int* x         = (const int*)d_in[0];
  const int* batch     = (const int*)d_in[1];
  const int* src       = (const int*)d_in[2];
  const int* dst       = (const int*)d_in[2] + EE;
  const float* snorm   = (const float*)d_in[3];
  const int* edge_attr = (const int*)d_in[4];
  const float* atom_emb = (const float*)d_in[5];
  const float* bond_emb = (const float*)d_in[6];
  const float* pre_W  = (const float*)d_in[7];
  const float* pre_b  = (const float*)d_in[8];
  const float* post_W = (const float*)d_in[9];
  const float* post_b = (const float*)d_in[10];
  const float* mix_W  = (const float*)d_in[11];
  const float* mix_b  = (const float*)d_in[12];
  const float* bn_g   = (const float*)d_in[13];
  const float* bn_b   = (const float*)d_in[14];
  const float* rW1 = (const float*)d_in[15];
  const float* rb1 = (const float*)d_in[16];
  const float* rW2 = (const float*)d_in[17];
  const float* rb2 = (const float*)d_in[18];
  const float* rW3 = (const float*)d_in[19];
  const float* rb3 = (const float*)d_in[20];
  float* out = (float*)d_out;

  // carve workspace
  size_t off = 0;
  char* base = (char*)d_ws;
  auto take = [&](size_t bytes) -> char* {
    char* p = base + off;
    off += (bytes + 255) & ~(size_t)255;
    return p;
  };
  float*  h      = (float*)  take((size_t)NN * DD * 4);
  half_t* hf     = (half_t*) take((size_t)NN * DD * 2);
  half_t* ef     = (half_t*) take((size_t)EE * DD * 2);
  uint4*  aggbuf = (uint4*)  take((size_t)NN * DD * 16);   // interleaved {sum,sumsq,max,min}
  half_t* aggf   = (half_t*) take((size_t)NN * 512 * 2);
  half_t* ph1    = (half_t*) take((size_t)NN * DD * 2);
  float*  ph     = (float*)  take((size_t)NN * DD * 4);
  float*  deg    = (float*)  take((size_t)NN * 4);
  float*  amp    = (float*)  take((size_t)NN * 4);
  float*  att    = (float*)  take((size_t)NN * 4);
  float*  colsum = (float*)  take(128 * 4);
  float*  colsq  = (float*)  take(128 * 4);
  half_t* preWh  = (half_t*) take((size_t)LL * 384 * 128 * 2);
  half_t* postWh = (half_t*) take((size_t)LL * 1664 * 128 * 2);
  half_t* mixWh  = (half_t*) take((size_t)LL * 128 * 128 * 2);
  float*  gsum   = (float*)  take((size_t)GG * DD * 4);
  float*  gcnt   = (float*)  take((size_t)GG * 4);

  const int T = 256;
  // pack weights straight into WMMA B-fragment order (one contiguous v16h per lane)
  pack_w_kernel<<<cdiv(LL * 384 * 128, T), T, 0, stream>>>(pre_W, preWh, 384, LL);
  pack_w_kernel<<<cdiv(LL * 1664 * 128, T), T, 0, stream>>>(post_W, postWh, 1664, LL);
  pack_w_kernel<<<cdiv(LL * 128 * 128, T), T, 0, stream>>>(mix_W, mixWh, 128, LL);
  // embeddings + degree scalers
  node_embed_kernel<<<cdiv(NN * DD, T), T, 0, stream>>>(x, atom_emb, h, hf);
  edge_embed_kernel<<<cdiv(EE * DD, T), T, 0, stream>>>(edge_attr, bond_emb, ef);
  fill_u32_kernel<<<cdiv(NN, T), T, 0, stream>>>((unsigned*)deg, 0u, NN);
  deg_kernel<<<cdiv(EE, T), T, 0, stream>>>(dst, deg);
  scaler_kernel<<<cdiv(NN, T), T, 0, stream>>>(deg, amp, att);

  for (int l = 0; l < LL; ++l) {
    agg_init_kernel<<<cdiv(NN * DD, T), T, 0, stream>>>(aggbuf, NN * DD);
    fill_u32_kernel<<<1, 128, 0, stream>>>((unsigned*)colsum, 0u, 128);
    fill_u32_kernel<<<1, 128, 0, stream>>>((unsigned*)colsq, 0u, 128);

    pretrans_agg_kernel<<<EE / 16, T, 0, stream>>>(
        hf, ef, preWh + (size_t)l * 384 * 128, pre_b + l * 128, src, dst, aggbuf);
    agg_finalize_kernel<<<cdiv(NN * DD, T), T, 0, stream>>>(aggbuf, deg, aggf);
    posttrans_kernel<<<NN / 16, T, 0, stream>>>(
        hf, aggf, postWh + (size_t)l * 1664 * 128, post_b + l * 128, amp, att, ph1);
    mix_kernel<<<NN / 16, T, 0, stream>>>(
        ph1, mixWh + (size_t)l * 128 * 128, mix_b + l * 128, snorm, ph, colsum, colsq);
    bn_residual_kernel<<<cdiv(NN * DD, T), T, 0, stream>>>(
        h, hf, ph, colsum, colsq, bn_g + l * 128, bn_b + l * 128);
  }

  fill_u32_kernel<<<cdiv(GG * DD, T), T, 0, stream>>>((unsigned*)gsum, 0u, GG * DD);
  fill_u32_kernel<<<1, GG, 0, stream>>>((unsigned*)gcnt, 0u, GG);
  pool_kernel<<<cdiv(NN * DD, T), T, 0, stream>>>(h, batch, gsum, gcnt);
  readout_kernel<<<GG, 128, 0, stream>>>(gsum, gcnt, rW1, rb1, rW2, rb2, rW3, rb3, out);
}
